// AttentionHead_48447231098912
// MI455X (gfx1250) — matrix operand order repaired and checked by reading, not compiled, not run on hardware
//
#include <hip/hip_runtime.h>

typedef __attribute__((ext_vector_type(16))) _Float16 v16h;
typedef __attribute__((ext_vector_type(8)))  float    v8f;
typedef __attribute__((ext_vector_type(4)))  unsigned int v4u;
typedef __attribute__((ext_vector_type(4)))  int      v4i;
typedef __attribute__((ext_vector_type(8)))  int      v8i;

#define DEVINL __device__ __forceinline__

constexpr int Bc = 4;
constexpr int Sc = 4096;
constexpr int Ec = 1024;
constexpr int Dc = 64;   // QD == VD == 64

DEVINL v8f wmma16(v16h a, v16h b, v8f c) {
  // v_wmma_f32_16x16x32_f16 : D = A(16x32 f16) * B(32x16 f16) + C(16x16 f32)
  return __builtin_amdgcn_wmma_f32_16x16x32_f16(false, a, false, b, (short)0, c,
                                                false, false);
}

// ---- WMMA operand loaders (contiguous k = 16h + s order, same for A and B, wave32) -------------
// A 16x32 (MxK) f16: lane m=l&15, k-half = (l>=16)?16:0; VGPR v holds
// k = khalf + 2v + {0,1}

DEVINL v16h loadA_f32(const float* src, int ld) {  // f32 source, cvt to f16
  const int l   = threadIdx.x & 31;
  const int row = l & 15;
  const int kb  = (l >> 4) << 4;
  v16h a;
#pragma unroll
  for (int v = 0; v < 8; ++v) {
    const int k = kb + 2 * v;
    const float2 f = *(const float2*)(src + (size_t)row * ld + k);
    a[2 * v]     = (_Float16)f.x;
    a[2 * v + 1] = (_Float16)f.y;
  }
  return a;
}

DEVINL v16h loadA_f16(const _Float16* src, int ld) {
  const int l   = threadIdx.x & 31;
  const int row = l & 15;
  const int kb  = (l >> 4) << 4;
  v16h a;
#pragma unroll
  for (int v = 0; v < 8; ++v) {
    const int k = kb + 2 * v;
    a[2 * v]     = src[(size_t)row * ld + k];
    a[2 * v + 1] = src[(size_t)row * ld + k + 1];
  }
  return a;
}

// B 32x16 (KxN) f16: lane n=l&15, k-half = (l>=16)?16:0; VGPR v holds
// k = khalf + 2v + {0,1}.
// "_T" variant: element (k,n) comes from src[n*ld + k]  (e.g. W[n][e], K^T)
DEVINL v16h loadB_T_f32(const float* src, int ld) {
  const int l  = threadIdx.x & 31;
  const int n  = l & 15;
  const int kh = (l >> 4) << 4;
  v16h b;
#pragma unroll
  for (int v = 0; v < 8; ++v) {
    const int k = kh + 2 * v;
    const float2 f = *(const float2*)(src + (size_t)n * ld + k);
    b[2 * v]     = (_Float16)f.x;
    b[2 * v + 1] = (_Float16)f.y;
  }
  return b;
}

DEVINL v16h loadB_T_f16(const _Float16* src, int ld) {
  const int l  = threadIdx.x & 31;
  const int n  = l & 15;
  const int kh = (l >> 4) << 4;
  v16h b;
#pragma unroll
  for (int v = 0; v < 8; ++v) {
    const int k = kh + 2 * v;
    b[2 * v]     = src[(size_t)n * ld + k];
    b[2 * v + 1] = src[(size_t)n * ld + k + 1];
  }
  return b;
}

// "_N" variant: element (k,n) comes from src[k*ld + n]  (e.g. V[key][d])
DEVINL v16h loadB_N_f16(const _Float16* src, int ld) {
  const int l  = threadIdx.x & 31;
  const int n  = l & 15;
  const int kh = (l >> 4) << 4;
  v16h b;
#pragma unroll
  for (int v = 0; v < 8; ++v) {
    const int k = kh + 2 * v;
    b[2 * v]     = src[(size_t)k * ld + n];
    b[2 * v + 1] = src[(size_t)(k + 1) * ld + n];
  }
  return b;
}

// ---- Tensor Data Mover: DMA a [tile_rows x row_elems] f16 tile ----------
// D# per CDNA5 ISA. 2-D tensor, no padding/iterate.
DEVINL void tdm_load_tile_f16(const _Float16* gsrc, unsigned lds_addr,
                              int tile_rows, int row_elems) {
  const unsigned long long ga = (unsigned long long)(uintptr_t)gsrc;
  v4u g0;
  g0[0] = 1u;                      // [1:0] count=1 (valid user descriptor)
  g0[1] = lds_addr;                // [63:32]  lds_addr (bytes)
  g0[2] = (unsigned)ga;            // [95:64]  global_addr low
  g0[3] = ((unsigned)(ga >> 32) & 0x01FFFFFFu)  // [120:96] global_addr high
        | (2u << 30);              // [127:126] type = 2 ("image")
  v8i g1;
  g1[0] = 1 << 16;                 // data_size=1 (2 bytes); wg_mask=0
  g1[1] = row_elems << 16;         // tensor_dim0 low16 (bits 79:48)
  g1[2] = tile_rows << 16;         // tensor_dim1 low16 (bits 111:80)
  g1[3] = row_elems << 16;         // tile_dim0       (bits 127:112)
  g1[4] = tile_rows;               // tile_dim1       (bits 143:128), tile_dim2=0
  g1[5] = row_elems;               // tensor_dim0_stride low32 (bits 191:160)
  g1[6] = 0;                       // stride high / tensor_dim1_stride
  g1[7] = 0;
  const v4i gz = {0, 0, 0, 0};
#if defined(__clang_major__) && (__clang_major__ >= 23)
  const v8i gz8 = {0, 0, 0, 0, 0, 0, 0, 0};
  __builtin_amdgcn_tensor_load_to_lds(g0, g1, gz, gz, gz8, 0);
#else
  __builtin_amdgcn_tensor_load_to_lds(g0, g1, gz, gz, 0);
#endif
}

// Hide the LDS pointer from the optimizer: sK/sV are only written by the TDM,
// which the compiler cannot see; this keeps the operand loads alive.
DEVINL const _Float16* lds_launder(const _Float16* p) {
  asm volatile("" : "+v"(p) : : "memory");
  return p;
}

// ---- Phase 1: x[16384x1024]f32 @ W^T[1024x64] + bias -> f16 -------------
// One wave per 16x64 output tile; 8 waves/block. Streams x once (mem-bound).
__global__ __launch_bounds__(256) void proj_qkv(const float* __restrict__ x,
                                                const float* __restrict__ W,
                                                const float* __restrict__ bias,
                                                _Float16* __restrict__ dst) {
  const int w    = threadIdx.x >> 5;
  const int lane = threadIdx.x & 31;
  const int row0 = blockIdx.x * 128 + w * 16;

  v8f acc[4] = {};
  for (int e0 = 0; e0 < Ec; e0 += 32) {
    const v16h a = loadA_f32(x + (size_t)row0 * Ec + e0, Ec);
#pragma unroll
    for (int j = 0; j < 4; ++j) {
      const v16h bt = loadB_T_f32(W + (size_t)(j * 16) * Ec + e0, Ec);
      acc[j] = wmma16(a, bt, acc[j]);
    }
  }

  const int ncol  = lane & 15;
  const int rbase = (lane >> 4) << 3;
#pragma unroll
  for (int j = 0; j < 4; ++j) {
    const float bj = bias[j * 16 + ncol];
#pragma unroll
    for (int r = 0; r < 8; ++r) {
      dst[(size_t)(row0 + rbase + r) * Dc + j * 16 + ncol] =
          (_Float16)(acc[j][r] + bj);
    }
  }
}

// ---- Phase 2: causal flash attention over f16 q/k/v ---------------------
// Block = 4 waves, 64 query rows; wave owns a 16-row tile, Q held in regs.
// 64-key K/V chunks double-buffered in LDS via TDM: chunk i+1 streams in
// while all 4 waves compute on chunk i (TENSORcnt in-order completion).
__global__ __launch_bounds__(128) void attn_fused(
    const _Float16* __restrict__ qf, const _Float16* __restrict__ kf,
    const _Float16* __restrict__ vf, float* __restrict__ out) {
  __shared__ _Float16 sK[2][64][64];   // 2 x 8 KB, written only by TDM
  __shared__ _Float16 sV[2][64][64];   // 2 x 8 KB, written only by TDM
  __shared__ _Float16 sP[4][16][68];   // per-wave P transpose staging

  const int b     = blockIdx.y;
  const int q0blk = blockIdx.x * 64;
  const int w     = threadIdx.x >> 5;
  const int lane  = threadIdx.x & 31;
  const int ncol  = lane & 15;
  const int rbase = (lane >> 4) << 3;
  const int q0    = q0blk + w * 16;

  const _Float16* qb  = qf + (size_t)b * Sc * Dc;
  const _Float16* kbp = kf + (size_t)b * Sc * Dc;
  const _Float16* vbp = vf + (size_t)b * Sc * Dc;

  const v16h aq0 = loadA_f16(qb + (size_t)q0 * Dc, Dc);       // K = 0..31
  const v16h aq1 = loadA_f16(qb + (size_t)q0 * Dc + 32, Dc);  // K = 32..63

  v8f o[4] = {};
  float m[8], lsum[8];
#pragma unroll
  for (int r = 0; r < 8; ++r) { m[r] = -1e30f; lsum[r] = 0.0f; }

  const unsigned sKl[2] = {(unsigned)(uintptr_t)&sK[0][0][0],
                           (unsigned)(uintptr_t)&sK[1][0][0]};
  const unsigned sVl[2] = {(unsigned)(uintptr_t)&sV[0][0][0],
                           (unsigned)(uintptr_t)&sV[1][0][0]};
  const _Float16* sKp[2] = {lds_launder(&sK[0][0][0]),
                            lds_launder(&sK[1][0][0])};
  const _Float16* sVp[2] = {lds_launder(&sV[0][0][0]),
                            lds_launder(&sV[1][0][0])};

  const int kend = q0blk + 64;  // causal: keys < q0blk+64 only

  // Prologue: kick off chunk 0 into buffer 0.
  if (w == 0) {
    tdm_load_tile_f16(kbp, sKl[0], 64, Dc);
    tdm_load_tile_f16(vbp, sVl[0], 64, Dc);
  }

  int buf = 0;
  for (int kb0 = 0; kb0 < kend; kb0 += 64, buf ^= 1) {
    if (w == 0) {
      const int knext = kb0 + 64;
      if (knext < kend) {
        // Prefetch chunk i+1 into the other buffer, then wait until only
        // those two DMAs remain outstanding -> chunk i is resident.
        tdm_load_tile_f16(kbp + (size_t)knext * Dc, sKl[buf ^ 1], 64, Dc);
        tdm_load_tile_f16(vbp + (size_t)knext * Dc, sVl[buf ^ 1], 64, Dc);
        __builtin_amdgcn_s_wait_tensorcnt(2);
      } else {
        __builtin_amdgcn_s_wait_tensorcnt(0);
      }
      asm volatile("" ::: "memory");
    }
    __syncthreads();

    const _Float16* sKb = sKp[buf];
    const _Float16* sVb = sVp[buf];

    // S = Q(16x64) @ K^T(64x64) -> four 16x16 f32 tiles
    v8f c[4] = {};
#pragma unroll
    for (int j = 0; j < 4; ++j) {
      v16h bt = loadB_T_f16(sKb + (size_t)(j * 16) * Dc, Dc);
      c[j] = wmma16(aq0, bt, c[j]);
      bt = loadB_T_f16(sKb + (size_t)(j * 16) * Dc + 32, Dc);
      c[j] = wmma16(aq1, bt, c[j]);
    }

    // Online softmax. C-layout: row = rbase+r (per VGPR), col = ncol
    // (per lane); row reductions are 16-lane shuffles.
#pragma unroll
    for (int r = 0; r < 8; ++r) {
      const int qi = q0 + rbase + r;
      float s[4];
      float mr = -1e30f;
#pragma unroll
      for (int j = 0; j < 4; ++j) {
        s[j] = c[j][r] * 0.125f;  // 1/sqrt(64)
        if (kb0 + j * 16 + ncol > qi) s[j] = -1e30f;
        mr = fmaxf(mr, s[j]);
      }
#pragma unroll
      for (int off = 8; off; off >>= 1)
        mr = fmaxf(mr, __shfl_xor(mr, off, 32));
      const float mn   = fmaxf(m[r], mr);
      const float corr = __expf(m[r] - mn);
      float rs = 0.0f;
#pragma unroll
      for (int j = 0; j < 4; ++j) {
        const float p = __expf(s[j] - mn);
        rs += p;
        sP[w][rbase + r][j * 16 + ncol] = (_Float16)p;  // C-layout -> LDS
      }
#pragma unroll
      for (int off = 8; off; off >>= 1) rs += __shfl_xor(rs, off, 32);
      lsum[r] = lsum[r] * corr + rs;
      m[r]    = mn;
#pragma unroll
      for (int j = 0; j < 4; ++j) o[j][r] *= corr;
    }

    // O += P(16x64) @ V(64x64); P re-read from LDS in A-layout.
    const v16h ap0 = loadA_f16(&sP[w][0][0], 68);   // keys 0..31
    const v16h ap1 = loadA_f16(&sP[w][0][32], 68);  // keys 32..63
#pragma unroll
    for (int j = 0; j < 4; ++j) {
      v16h bv = loadB_N_f16(sVb + j * 16, Dc);                   // k 0..31
      o[j] = wmma16(ap0, bv, o[j]);
      bv = loadB_N_f16(sVb + (size_t)32 * Dc + j * 16, Dc);      // k 32..63
      o[j] = wmma16(ap1, bv, o[j]);
    }
    // All waves done reading `buf` before its DMA is re-issued next iter.
    __syncthreads();
  }

  float* ob = out + ((size_t)b * Sc + q0) * Dc;
#pragma unroll
  for (int r = 0; r < 8; ++r) {
    const float inv = 1.0f / lsum[r];
    float* orow = ob + (size_t)(rbase + r) * Dc;
#pragma unroll
    for (int j = 0; j < 4; ++j) orow[j * 16 + ncol] = o[j][r] * inv;
  }
}

extern "C" void kernel_launch(void* const* d_in, const int* in_sizes, int n_in,
                              void* d_out, int out_size, void* d_ws,
                              size_t ws_size, hipStream_t stream) {
  const float* query = (const float*)d_in[0];
  const float* key   = (const float*)d_in[1];
  const float* value = (const float*)d_in[2];
  const float* Wq    = (const float*)d_in[3];
  const float* bq    = (const float*)d_in[4];
  const float* Wk    = (const float*)d_in[5];
  const float* bk    = (const float*)d_in[6];
  const float* Wv    = (const float*)d_in[7];
  const float* bv    = (const float*)d_in[8];
  float* out = (float*)d_out;

  _Float16* qf = (_Float16*)d_ws;                 // 2 MB each, 6 MB total
  _Float16* kf = qf + (size_t)Bc * Sc * Dc;
  _Float16* vf = kf + (size_t)Bc * Sc * Dc;

  dim3 g1((Bc * Sc) / 128);  // 128 blocks x 8 waves, 16 rows/wave
  proj_qkv<<<g1, 256, 0, stream>>>(query, Wq, bq, qf);
  proj_qkv<<<g1, 256, 0, stream>>>(key,   Wk, bk, kf);
  proj_qkv<<<g1, 256, 0, stream>>>(value, Wv, bv, vf);

  dim3 g2(Sc / 64, Bc);      // 64 q-tiles x 4 batches, 4 waves/block
  attn_fused<<<g2, 128, 0, stream>>>(qf, kf, vf, out);

  (void)in_sizes; (void)n_in; (void)out_size; (void)ws_size;
}
